// Clusteror_30889404793414
// MI455X (gfx1250) — matrix operand order repaired and checked by reading, not compiled, not run on hardware
//
#include <hip/hip_runtime.h>

typedef unsigned short u16;
typedef unsigned int   u32;
typedef unsigned long long u64;

typedef __bf16 bf16_t;
typedef bf16_t v16bf __attribute__((ext_vector_type(16)));
typedef float  v8f   __attribute__((ext_vector_type(8)));
typedef u32    v4u   __attribute__((ext_vector_type(4)));
typedef int    v4i   __attribute__((ext_vector_type(4)));
typedef int    v8i   __attribute__((ext_vector_type(8)));

#define D_FEAT 512
#define P_VN   64
#define D_ATT  64

#if defined(__has_builtin)
#  if __has_builtin(__builtin_amdgcn_tensor_load_to_lds) && __has_builtin(__builtin_amdgcn_s_wait_tensorcnt)
#    define HAVE_TDM 1
#  endif
#endif
#ifndef HAVE_TDM
#  define HAVE_TDM 0
#endif

// ---------------- bf16 helpers ----------------
__device__ inline u16 f2b(float f) {
  u32 u = __builtin_bit_cast(u32, f);
  u += 0x7fffu + ((u >> 16) & 1u);          // round-to-nearest-even
  return (u16)(u >> 16);
}
__device__ inline float b2f(u16 s) {
  u32 u = ((u32)s) << 16;
  return __builtin_bit_cast(float, u);
}

// ---------------- elementwise conversion kernels ----------------
__global__ void k_f32_to_bf16(const float* __restrict__ in, u16* __restrict__ out, size_t n) {
  size_t i = (size_t)blockIdx.x * blockDim.x + threadIdx.x;
  if (i < n) out[i] = f2b(in[i]);
}

__global__ void k_pack_x(const float* __restrict__ x, const float* __restrict__ vemb,
                         u16* __restrict__ out, int N, int M) {
  size_t i = (size_t)blockIdx.x * blockDim.x + threadIdx.x;
  size_t n = (size_t)M * D_FEAT;
  if (i >= n) return;
  int row = (int)(i >> 9);
  int c   = (int)(i & 511);
  float v = (row < N) ? x[i] : vemb[(size_t)(row - N) * D_FEAT + c];
  out[i] = f2b(v);
}

__global__ void k_add_vbias_bf16(u16* __restrict__ buf, const float* __restrict__ vb, int n) {
  int i = blockIdx.x * blockDim.x + threadIdx.x;
  if (i < n) buf[i] = f2b(b2f(buf[i]) + vb[i]);
}

__global__ void k_set_loss(float* __restrict__ out, size_t idx) {
  if (threadIdx.x == 0 && blockIdx.x == 0) out[idx] = 0.0f;
}

// ---------------- WMMA bf16 GEMM:  C[m,n] = sum_k A[m,k] * W[n,k] + bias[n] ----------------
// Block tile 128(M) x 128(N), 256 threads = 8 waves, wave tile 32x64 (2x4 of 16x16 WMMA).
// K staged in steps of 32 through double-buffered LDS (stride padded to 40 u16).
// USE_TDM=1: tiles DMA'd by the Tensor Data Mover (tensor_load_to_lds), hardware LDS
// padding (pad_interval=16 DW, pad_amount=4 DW -> 40-u16 stride), OOB rows zero-filled
// via tensor_dim1, completion via s_wait_tensorcnt. USE_TDM=0: register staging.
#define BM 128
#define BN 128
#define BK 32
#define LDS_STRIDE 40

__device__ inline v16bf load_frag_A(const u16* s, int kb) {
  union { v16bf v; uint4 q[2]; } u;
  u.q[0] = *(const uint4*)(s + kb);        // K = kb .. kb+7
  u.q[1] = *(const uint4*)(s + kb + 8);    // K = kb+8 .. kb+15
  return u.v;
}
__device__ inline v16bf load_frag_B(const u16* s, int kbase) {
  union { v16bf v; uint4 q[2]; } u;
  u.q[0] = *(const uint4*)(s + kbase);     // K = kbase .. kbase+7
  u.q[1] = *(const uint4*)(s + kbase + 8); // K = kbase+8 .. kbase+15
  return u.v;
}

#if HAVE_TDM
// 2D TDM tile load: 128 rows x 32 bf16 from row-major (strideElems) into LDS at lds_addr,
// LDS row stride padded to 40 u16 by the TDM pad unit. Rows >= tdim1 read as zero.
__device__ inline void tdm_load_tile(u32 lds_addr, const u16* gptr, u32 tdim1, u64 strideElems) {
  u64 ga = (u64)(uintptr_t)gptr;
  v4u g0;
  g0.x = 1u;                                            // count=1, user descriptor
  g0.y = lds_addr;                                      // lds_addr (bytes)
  g0.z = (u32)ga;                                       // global_addr[31:0]
  g0.w = (u32)((ga >> 32) & 0x01FFFFFFu) | (2u << 30);  // global_addr[56:32] | type=2
  const u32 tdim0 = 32u;
  v8i g1;
  g1[0] = (int)((1u << 16)      // data_size = 2 bytes
              | (1u << 20)      // pad_enable
              | (3u << 22)      // pad_interval: 16 DWORDs (one 32-elem bf16 row)
              | (3u << 25));    // pad_amount: 4 DWORDs (8 u16) -> 40-u16 LDS stride
  g1[1] = (int)((tdim0 & 0xFFFFu) << 16);                       // tensor_dim0[15:0]
  g1[2] = (int)((tdim0 >> 16) | ((tdim1 & 0xFFFFu) << 16));     // tensor_dim0[31:16] | tensor_dim1[15:0]
  g1[3] = (int)((tdim1 >> 16) | (32u << 16));                   // tensor_dim1[31:16] | tile_dim0=32
  g1[4] = (int)128u;                                            // tile_dim1=128, tile_dim2=0
  g1[5] = (int)(u32)strideElems;                                // tensor_dim0_stride[31:0]
  g1[6] = (int)((u32)(strideElems >> 32) & 0xFFFFu);            // tensor_dim0_stride[47:32]
  g1[7] = 0;
  v4i z4 = {0, 0, 0, 0};
#if __clang_major__ >= 23
  v8i z8 = {0, 0, 0, 0, 0, 0, 0, 0};
  __builtin_amdgcn_tensor_load_to_lds(g0, g1, z4, z4, z8, 0);
#else
  __builtin_amdgcn_tensor_load_to_lds(g0, g1, z4, z4, 0);
#endif
}
#endif

template <int USE_TDM, int GATHER>
__global__ __launch_bounds__(256) void k_gemm_bf16(
    const u16* __restrict__ A, int physK,        // A physical row length (elems)
    const u16* __restrict__ W,                   // (NC x K) row-major, bf16
    const float* __restrict__ bias,              // (NC)
    float* __restrict__ Cmain, float* __restrict__ Crep, int splitRow,
    int M, int K, int NC,
    const int* __restrict__ cidx) {

  __shared__ __align__(16) u16 As[2][BM * LDS_STRIDE];
  __shared__ __align__(16) u16 Bs[2][BN * LDS_STRIDE];

  const int tid  = threadIdx.x;
  const int lane = tid & 31;
  const int wave = tid >> 5;
  const int wm   = wave & 3;    // 0..3 -> M offset wm*32
  const int wn   = wave >> 2;   // 0..1 -> N offset wn*64
  const int half = lane >> 4;   // 0/1
  const int l16  = lane & 15;

  const int blockM = blockIdx.x * BM;
  const int blockN = blockIdx.y * BN;

  // ---- manual staging setup (fallback + gather path)
  const int srow  = tid >> 1;         // 0..127
  const int spart = tid & 1;          // elems [spart*16, spart*16+7]
  const u16 *aBase0 = nullptr, *aBase1 = nullptr, *bBase = nullptr;
  u16 *aDst0 = nullptr, *aDst1 = nullptr, *bDst0 = nullptr, *bDst1 = nullptr;
  if constexpr (!USE_TDM) {
    int aRow = blockM + srow;
    if (aRow > M - 1) aRow = M - 1;   // clamp: garbage only feeds unstored C rows
    int grow = aRow;
    if (GATHER && aRow < splitRow) grow = splitRow + cidx[aRow];
    aBase0 = A + (size_t)aRow * physK + spart * 16;            // k0 <  physK
    aBase1 = A + (size_t)grow * physK - physK + spart * 16;    // k0 >= physK (gather)
    bBase  = W + (size_t)(blockN + srow) * K + spart * 16;
    aDst0 = &As[0][srow * LDS_STRIDE + spart * 16];
    aDst1 = &As[1][srow * LDS_STRIDE + spart * 16];
    bDst0 = &Bs[0][srow * LDS_STRIDE + spart * 16];
    bDst1 = &Bs[1][srow * LDS_STRIDE + spart * 16];
  }

  v8f acc[2][4];
  {
    v8f z = {0.f, 0.f, 0.f, 0.f, 0.f, 0.f, 0.f, 0.f};
#pragma unroll
    for (int i = 0; i < 2; ++i)
#pragma unroll
      for (int j = 0; j < 4; ++j) acc[i][j] = z;
  }

  // ---- prologue: stage k0 = 0 into buffer 0
#if HAVE_TDM
  u32 ldsA[2], ldsB[2];
  if constexpr (USE_TDM) {
    ldsA[0] = (u32)(uintptr_t)&As[0][0];
    ldsA[1] = (u32)(uintptr_t)&As[1][0];
    ldsB[0] = (u32)(uintptr_t)&Bs[0][0];
    ldsB[1] = (u32)(uintptr_t)&Bs[1][0];
    if (wave == 0) {
      tdm_load_tile(ldsA[0], A + (size_t)blockM * physK, (u32)(M - blockM), (u64)physK);
      tdm_load_tile(ldsB[0], W + (size_t)blockN * K,     (u32)(NC - blockN), (u64)K);
      __builtin_amdgcn_s_wait_tensorcnt(0);
    }
  }
#endif
  if constexpr (!USE_TDM) {
    ((uint4*)aDst0)[0] = ((const uint4*)aBase0)[0];    // k0 = 0 always < physK
    ((uint4*)bDst0)[0] = ((const uint4*)bBase)[0];
  }
  __syncthreads();

  const int nIter = K / BK;
  const int kbA = half * 16;    // A: half-wave contiguous 16
  const int kbB = half * 16;    // B: half-wave contiguous 16

  for (int it = 0; it < nIter; ++it) {
    const int cur = it & 1;
    const int nxt = cur ^ 1;
    const int k0n = (it + 1) * BK;
    const bool haveNext = (k0n < K);

    uint4 rga, rgb;
    if constexpr (USE_TDM) {
#if HAVE_TDM
      if (haveNext && wave == 0) {
        tdm_load_tile(ldsA[nxt], A + (size_t)blockM * physK + k0n, (u32)(M - blockM), (u64)physK);
        tdm_load_tile(ldsB[nxt], W + (size_t)blockN * K + k0n,     (u32)(NC - blockN), (u64)K);
      }
#endif
    } else {
      if (haveNext) {
        const u16* sa = (k0n < physK ? aBase0 : aBase1) + k0n;
        const u16* sb = bBase + k0n;
        rga = ((const uint4*)sa)[0];
        rgb = ((const uint4*)sb)[0];
        __builtin_prefetch(sa + BK, 0, 1);          // global_prefetch_b8
        __builtin_prefetch(sb + BK, 0, 1);
      }
    }

    // ---- compute from buffer `cur` (TDM copy of `nxt` overlaps these WMMAs)
    v16bf afrag[2], bfrag[4];
#pragma unroll
    for (int i = 0; i < 2; ++i)
      afrag[i] = load_frag_A(&As[cur][(wm * 32 + i * 16 + l16) * LDS_STRIDE], kbA);
#pragma unroll
    for (int j = 0; j < 4; ++j)
      bfrag[j] = load_frag_B(&Bs[cur][(wn * 64 + j * 16 + l16) * LDS_STRIDE], kbB);

#pragma unroll
    for (int i = 0; i < 2; ++i)
#pragma unroll
      for (int j = 0; j < 4; ++j)
        acc[i][j] = __builtin_amdgcn_wmma_f32_16x16x32_bf16(
            false, afrag[i], false, bfrag[j], (short)0, acc[i][j], false, false);

    // ---- finish staging of `nxt` before the barrier
    if constexpr (USE_TDM) {
#if HAVE_TDM
      if (haveNext && wave == 0) __builtin_amdgcn_s_wait_tensorcnt(0);
#endif
    } else {
      if (haveNext) {
        if (cur) { ((uint4*)aDst0)[0] = rga; ((uint4*)bDst0)[0] = rgb; }
        else     { ((uint4*)aDst1)[0] = rga; ((uint4*)bDst1)[0] = rgb; }
      }
    }
    __syncthreads();
  }

  // ---- store: C/D layout -> VGPR r: lanes0-15 M=r, lanes16-31 M=r+8; N = lane&15
#pragma unroll
  for (int i = 0; i < 2; ++i) {
#pragma unroll
    for (int j = 0; j < 4; ++j) {
      const int cbase = blockN + wn * 64 + j * 16 + l16;
      const float bi = bias[cbase];
#pragma unroll
      for (int r = 0; r < 8; ++r) {
        int row = blockM + wm * 32 + i * 16 + r + half * 8;
        if (row < M) {
          float v = acc[i][j][r] + bi;
          if (Crep && row >= splitRow)
            Crep[(size_t)(row - splitRow) * NC + cbase] = v;
          else
            Cmain[(size_t)row * NC + cbase] = v;
        }
      }
    }
  }
}

// ---------------- fused LayerNorm + ELU (+optional vnode bias) -> bf16 ----------------
__global__ __launch_bounds__(256) void k_ln_elu(
    const float* __restrict__ Hf, const float* __restrict__ g,
    const float* __restrict__ beta, const float* __restrict__ vbias,
    u16* __restrict__ out, int M, int splitRow) {
  int row  = blockIdx.x * 8 + (threadIdx.x >> 5);
  int lane = threadIdx.x & 31;
  if (row >= M) return;

  const float4* xv = (const float4*)(Hf + (size_t)row * D_FEAT);
  float4 v[4];
  float sum = 0.f, ss = 0.f;
#pragma unroll
  for (int c = 0; c < 4; ++c) {
    v[c] = xv[c * 32 + lane];
    sum += v[c].x + v[c].y + v[c].z + v[c].w;
    ss  += v[c].x * v[c].x + v[c].y * v[c].y + v[c].z * v[c].z + v[c].w * v[c].w;
  }
#pragma unroll
  for (int o = 16; o > 0; o >>= 1) {
    sum += __shfl_xor(sum, o);
    ss  += __shfl_xor(ss, o);
  }
  const float mean = sum * (1.0f / D_FEAT);
  const float var  = ss * (1.0f / D_FEAT) - mean * mean;
  const float rstd = rsqrtf(var + 1e-5f);

  const bool addv = (vbias != nullptr) && (row >= splitRow);
  const float4* vb4 = addv ? (const float4*)(vbias + (size_t)(row - splitRow) * D_FEAT) : nullptr;

#pragma unroll
  for (int c = 0; c < 4; ++c) {
    int idx4 = c * 32 + lane;
    float4 g4 = ((const float4*)g)[idx4];
    float4 b4 = ((const float4*)beta)[idx4];
    float t0 = (v[c].x - mean) * rstd * g4.x + b4.x;
    float t1 = (v[c].y - mean) * rstd * g4.y + b4.y;
    float t2 = (v[c].z - mean) * rstd * g4.z + b4.z;
    float t3 = (v[c].w - mean) * rstd * g4.w + b4.w;
    t0 = t0 > 0.f ? t0 : expm1f(t0);
    t1 = t1 > 0.f ? t1 : expm1f(t1);
    t2 = t2 > 0.f ? t2 : expm1f(t2);
    t3 = t3 > 0.f ? t3 : expm1f(t3);
    if (addv) {
      float4 vb = vb4[idx4];
      t0 += vb.x; t1 += vb.y; t2 += vb.z; t3 += vb.w;
    }
    u32 lo = (u32)f2b(t0) | ((u32)f2b(t1) << 16);
    u32 hi = (u32)f2b(t2) | ((u32)f2b(t3) << 16);
    *(uint2*)(out + (size_t)row * D_FEAT + (size_t)idx4 * 4) = make_uint2(lo, hi);
  }
}

// ---------------- attention: dst = h[-P:] @ Wd^T + bd + attn_bias  (64x64) ----------------
__global__ void k_attn_dst(const u16* __restrict__ Hb, int N,
                           const float* __restrict__ Wd, const float* __restrict__ bd,
                           const float* __restrict__ attn_bias, float* __restrict__ dstb) {
  int gid = blockIdx.x * blockDim.x + threadIdx.x;
  if (gid >= P_VN * D_ATT) return;
  int p = gid >> 6, a = gid & 63;
  const u16* h = Hb + (size_t)(N + p) * D_FEAT;
  const float* w = Wd + (size_t)a * D_FEAT;
  float s = 0.f;
  for (int k = 0; k < D_FEAT; ++k) s += b2f(h[k]) * w[k];
  dstb[gid] = s + bd[a] + attn_bias[gid];
}

// ---------------- attention: per-row src, scores, leaky-relu, argmax ----------------
__global__ __launch_bounds__(64) void k_attn_argmax(
    const u16* __restrict__ Hb, const float* __restrict__ Ws, const float* __restrict__ bs,
    const float* __restrict__ dstb, int* __restrict__ cidx, float* __restrict__ map_out) {
  __shared__ float hrow[D_FEAT];
  __shared__ float srcv[D_ATT];
  __shared__ float red[D_ATT];
  __shared__ int   redi[D_ATT];
  const int row = blockIdx.x;
  const int t   = threadIdx.x;

  const u16* h = Hb + (size_t)row * D_FEAT;
#pragma unroll
  for (int i = 0; i < 8; ++i) hrow[t * 8 + i] = b2f(h[t * 8 + i]);
  __syncthreads();

  const float* w = Ws + (size_t)t * D_FEAT;
  float s = bs[t];
  for (int k = 0; k < D_FEAT; ++k) s += hrow[k] * w[k];
  srcv[t] = s;
  __syncthreads();

  float sc = 0.f;
  const float* d = dstb + t * D_ATT;
  for (int a = 0; a < D_ATT; ++a) sc += srcv[a] * d[a];
  sc = sc > 0.f ? sc : 0.2f * sc;       // leaky_relu; monotonic -> argmax preserved
  red[t] = sc; redi[t] = t;
  __syncthreads();
  for (int off = 32; off > 0; off >>= 1) {
    if (t < off) {
      if (red[t + off] > red[t]) { red[t] = red[t + off]; redi[t] = redi[t + off]; }
    }
    __syncthreads();
  }
  if (t == 0) {
    cidx[row]    = redi[0];
    map_out[row] = (float)redi[0];
  }
}

// ---------------- launch ----------------
extern "C" void kernel_launch(void* const* d_in, const int* in_sizes, int n_in,
                              void* d_out, int out_size, void* d_ws, size_t ws_size,
                              hipStream_t stream) {
  const float* x          = (const float*)d_in[0];
  const float* vemb       = (const float*)d_in[1];
  const float* vb_hid     = (const float*)d_in[2];
  const float* vb_dcd     = (const float*)d_in[3];
  const float* W_in2hid   = (const float*)d_in[4];
  const float* b_in2hid   = (const float*)d_in[5];
  const float* g_hid      = (const float*)d_in[6];
  const float* beta_hid   = (const float*)d_in[7];
  const float* W_enc      = (const float*)d_in[8];
  const float* b_enc      = (const float*)d_in[9];
  const float* g_dcd      = (const float*)d_in[10];
  const float* beta_dcd   = (const float*)d_in[11];
  const float* Ws         = (const float*)d_in[12];
  const float* bs         = (const float*)d_in[13];
  const float* Wd         = (const float*)d_in[14];
  const float* bd         = (const float*)d_in[15];
  const float* attn_bias  = (const float*)d_in[16];
  const float* W_aggr     = (const float*)d_in[17];
  const float* b_aggr     = (const float*)d_in[18];
  const float* W_out      = (const float*)d_in[19];
  const float* b_out      = (const float*)d_in[20];
  float* out = (float*)d_out;

  const int D = D_FEAT, P = P_VN;
  const int M = in_sizes[0] / D;       // N + P
  const int N = M - P;

  size_t off = 0;
  char* ws = (char*)d_ws;
  auto alloc = [&](size_t bytes) -> void* {
    void* p = ws + off;
    off = (off + bytes + 255) & ~(size_t)255;
    return p;
  };
  u16*   bufA = (u16*)alloc((size_t)M * D * 2);
  u16*   bufB = (u16*)alloc((size_t)M * D * 2);
  float* bufF = (float*)alloc((size_t)M * D * 4);
  u16*   wb1  = (u16*)alloc((size_t)D * D * 2);
  u16*   wb2  = (u16*)alloc((size_t)D * D * 2);
  u16*   wbA  = (u16*)alloc((size_t)D * 2 * D * 2);
  u16*   wbO  = (u16*)alloc((size_t)D * D * 2);
  float* dstb = (float*)alloc((size_t)P_VN * D_ATT * 4);
  int*   cidx = (int*)alloc((size_t)M * 4);
  (void)ws_size; (void)n_in; (void)out_size;

  const size_t lossIdx = (size_t)N * D;
  float* repPtr = out + lossIdx + 1;
  float* mapPtr = out + lossIdx + 1 + (size_t)P * D;

  const dim3 gemmGrid((M + BM - 1) / BM, D / BN);
  const int lnGrid = (M + 7) / 8;
  constexpr int KTDM = HAVE_TDM;       // dense GEMMs use the Tensor Data Mover if present

  // 0) weights -> bf16
  {
    size_t n = (size_t)D * D;
    k_f32_to_bf16<<<(n + 255) / 256, 256, 0, stream>>>(W_in2hid, wb1, n);
    k_f32_to_bf16<<<(n + 255) / 256, 256, 0, stream>>>(W_enc, wb2, n);
    k_f32_to_bf16<<<(n + 255) / 256, 256, 0, stream>>>(W_out, wbO, n);
    size_t na = (size_t)D * 2 * D;
    k_f32_to_bf16<<<(na + 255) / 256, 256, 0, stream>>>(W_aggr, wbA, na);
  }
  // 1) x with vnode rows -> bf16
  {
    size_t n = (size_t)M * D;
    k_pack_x<<<(n + 255) / 256, 256, 0, stream>>>(x, vemb, bufA, N, M);
  }
  // 2) H1f = A0 @ W_in2hid^T + b
  k_gemm_bf16<KTDM, 0><<<gemmGrid, 256, 0, stream>>>(bufA, D, wb1, b_in2hid,
      bufF, nullptr, M, M, D, D, nullptr);
  // 3) H1b = elu(ln(H1f)); last P rows += vnode_bias_hid
  k_ln_elu<<<lnGrid, 256, 0, stream>>>(bufF, g_hid, beta_hid, vb_hid, bufB, M, N);
  // 4) H2f = H1b @ W_enc^T + b_enc
  k_gemm_bf16<KTDM, 0><<<gemmGrid, 256, 0, stream>>>(bufB, D, wb2, b_enc,
      bufF, nullptr, M, M, D, D, nullptr);
  // 5) H2b = elu(ln(H2f))
  k_ln_elu<<<lnGrid, 256, 0, stream>>>(bufF, g_dcd, beta_dcd, nullptr, bufA, M, M);
  // 6) attention dst (64x64), then per-row argmax -> cluster_idx + mapping output
  k_attn_dst<<<(P_VN * D_ATT + 255) / 256, 256, 0, stream>>>(bufA, N, Wd, bd, attn_bias, dstb);
  k_attn_argmax<<<N, 64, 0, stream>>>(bufA, Ws, bs, dstb, cidx, mapPtr);
  // 7) last P rows of H2b += vnode_bias_dcd
  k_add_vbias_bf16<<<(P * D + 255) / 256, 256, 0, stream>>>(bufA + (size_t)N * D, vb_dcd, P * D);
  // 8) H3f = [H2b | H2b[gather]] @ W_aggr^T + b_aggr   (K = 1024, gathered manual staging)
  k_gemm_bf16<0, 1><<<gemmGrid, 256, 0, stream>>>(bufA, D, wbA, b_aggr,
      bufF, nullptr, N, M, 2 * D, D, cidx);
  // 9) H3b = elu(ln(H3f))
  k_ln_elu<<<lnGrid, 256, 0, stream>>>(bufF, g_dcd, beta_dcd, nullptr, bufB, M, M);
  // 10) Y = H3b @ W_out^T + b_out -> split into out / cluster_reps
  k_gemm_bf16<KTDM, 0><<<gemmGrid, 256, 0, stream>>>(bufB, D, wbO, b_out,
      out, repPtr, N, M, D, D, nullptr);
  // 11) loss = 0
  k_set_loss<<<1, 32, 0, stream>>>(out, lossIdx);
}
